// ImageRetrievalRankingLoss_25185688224588
// MI455X (gfx1250) — matrix operand order repaired and checked by reading, not compiled, not run on hardware
//
#include <hip/hip_runtime.h>
#include <hip/hip_bf16.h>

// ---------------------------------------------------------------------------
// VSE hinge ranking loss, fused:  sum_ij max(0, 1 - <im_j,s_j> + <im_i,s_j>),
// diagonal zeroed.  Compute-bound (68.7 GFLOP vs 33.5 MB in) -> bf16x3-split
// WMMA GEMM with fused epilogue; the 256 MB score matrix never touches memory.
// GLOBAL_LOAD_ASYNC_TO_LDS_B128 DMA double-buffering (no VGPR staging -> no
// scratch spills), 128x64 block tile sized for 4 waves/SIMD.
// ---------------------------------------------------------------------------

typedef __attribute__((ext_vector_type(16))) __bf16 v16bf;
typedef __attribute__((ext_vector_type(8)))  float  v8f;

union U16frag { uint4 q[2]; v16bf v; };

#define D_DIM   512
#define BM      128
#define BN      64
#define KC      32
#define KSTEPS  (D_DIM / KC)
#define LDSS    40        // LDS row stride in bf16 elems (32 data + 8 pad = 80B)

// LDS stage layout (elements): A_hi | A_lo | B_hi | B_lo
#define A_HI_OFF  0
#define A_LO_OFF  (BM * LDSS)                      // 5120
#define B_HI_OFF  (2 * BM * LDSS)                  // 10240
#define B_LO_OFF  (2 * BM * LDSS + BN * LDSS)      // 12800
#define STAGE_EL  (2 * BM * LDSS + 2 * BN * LDSS)  // 15360 elems = 30720 B

#if __has_builtin(__builtin_amdgcn_global_load_async_to_lds_b128)
#define HAVE_ASYNC_LDS 1
#else
#define HAVE_ASYNC_LDS 0
#endif

// Builtin parameter types per the clang diagnostic:
//   (v4i addrspace(1)*, v4i addrspace(3)*, const int offset, const int cpol)
// with v4i = int __attribute__((vector_size(16))).
typedef int v4i __attribute__((__vector_size__(4 * sizeof(int))));
typedef __attribute__((address_space(1))) v4i* gas1_t;
typedef __attribute__((address_space(3))) v4i* las3_t;

static __device__ __forceinline__ void wait_async0() {
#if __has_builtin(__builtin_amdgcn_s_wait_asynccnt)
    __builtin_amdgcn_s_wait_asynccnt(0);
#else
    asm volatile("s_wait_asynccnt 0x0" ::: "memory");
#endif
}

static __device__ __forceinline__ unsigned short bf16_rne(float x) {
    union { float f; unsigned u; } c; c.f = x;
    unsigned u = c.u;
    u += 0x7FFFu + ((u >> 16) & 1u);          // round-to-nearest-even
    return (unsigned short)(u >> 16);
}
static __device__ __forceinline__ float bf16_f32(unsigned short h) {
    union { unsigned u; float f; } c; c.u = ((unsigned)h) << 16;
    return c.f;
}

// ---------------------------------------------------------------------------
// Kernel 1: per-row fp32 diag[j] = <im_j, s_j>, convert im/s to bf16 hi/lo
// planes, zero the scalar output.
// ---------------------------------------------------------------------------
__global__ void __launch_bounds__(128)
vse_prep(const float* __restrict__ im, const float* __restrict__ s,
         unsigned short* __restrict__ im_hi, unsigned short* __restrict__ im_lo,
         unsigned short* __restrict__ s_hi,  unsigned short* __restrict__ s_lo,
         float* __restrict__ diag, float* __restrict__ out) {
    const int row  = blockIdx.x;
    const int t    = threadIdx.x;
    const size_t base = (size_t)row * D_DIM + t * 4;

    const float4 a = *(const float4*)(im + base);
    const float4 b = *(const float4*)(s  + base);
    const float av[4] = {a.x, a.y, a.z, a.w};
    const float bv[4] = {b.x, b.y, b.z, b.w};

    unsigned short ah[4], al[4], bh[4], bl[4];
    float dotp = 0.0f;
#pragma unroll
    for (int e = 0; e < 4; ++e) {
        ah[e] = bf16_rne(av[e]);
        al[e] = bf16_rne(av[e] - bf16_f32(ah[e]));
        bh[e] = bf16_rne(bv[e]);
        bl[e] = bf16_rne(bv[e] - bf16_f32(bh[e]));
        dotp += av[e] * bv[e];
    }
    uint2 p;
    p.x = (unsigned)ah[0] | ((unsigned)ah[1] << 16);
    p.y = (unsigned)ah[2] | ((unsigned)ah[3] << 16);
    *(uint2*)(im_hi + base) = p;
    p.x = (unsigned)al[0] | ((unsigned)al[1] << 16);
    p.y = (unsigned)al[2] | ((unsigned)al[3] << 16);
    *(uint2*)(im_lo + base) = p;
    p.x = (unsigned)bh[0] | ((unsigned)bh[1] << 16);
    p.y = (unsigned)bh[2] | ((unsigned)bh[3] << 16);
    *(uint2*)(s_hi + base) = p;
    p.x = (unsigned)bl[0] | ((unsigned)bl[1] << 16);
    p.y = (unsigned)bl[2] | ((unsigned)bl[3] << 16);
    *(uint2*)(s_lo + base) = p;

    __shared__ float red[128];
    red[t] = dotp;
    __syncthreads();
#pragma unroll
    for (int off = 64; off > 0; off >>= 1) {
        if (t < off) red[t] += red[t + off];
        __syncthreads();
    }
    if (t == 0) {
        diag[row] = red[0];
        if (row == 0) *out = 0.0f;     // init scalar accumulator
    }
}

// ---------------------------------------------------------------------------
// Kernel 2: 128x64 block-tile bf16x3 WMMA GEMM with fused hinge epilogue.
// 256 threads = 8 waves (wave32), waves 4(m) x 2(n), wave tile 32x32.
// Async DMA double buffering when the toolchain exposes the builtin.
// ---------------------------------------------------------------------------
__global__ void __launch_bounds__(256)
vse_gemm_hinge(const unsigned short* __restrict__ Ahi, const unsigned short* __restrict__ Alo,
               const unsigned short* __restrict__ Bhi, const unsigned short* __restrict__ Blo,
               const float* __restrict__ diag, float* __restrict__ out) {
#if HAVE_ASYNC_LDS
    __shared__ __align__(16) unsigned short lds[2][STAGE_EL];   // 60 KB
#else
    __shared__ __align__(16) unsigned short lds[1][STAGE_EL];   // 30 KB
#endif

    const int t    = threadIdx.x;
    const int lane = t & 31;
    const int w    = t >> 5;            // wave id 0..7
    const int wm   = (w & 3) * 32;      // wave m-offset in block tile
    const int wn   = (w >> 2) * 32;     // wave n-offset in block tile
    const int iBase = blockIdx.y * BM;
    const int jBase = blockIdx.x * BN;

    // Staging assignment: A planes -> thread t copies one 32B chunk (2x b128),
    // B planes -> one 16B chunk.  All 16B-aligned.
    const int arow = t >> 1;            // 0..127
    const int aseg = (t & 1) * 16;      // k-offset 0 or 16
    const int brow = t >> 2;            // 0..63
    const int bseg = (t & 3) * 8;       // k-offset 0,8,16,24
    const size_t aoff = (size_t)(iBase + arow) * D_DIM + aseg;
    const size_t boff = (size_t)(jBase + brow) * D_DIM + bseg;
    const int aldso = arow * LDSS + aseg;
    const int bldso = brow * LDSS + bseg;

#if HAVE_ASYNC_LDS
    auto issue_stage = [&](int ks, int buf) {
        const size_t ko = (size_t)ks * KC;
        unsigned short* l = lds[buf];
        __builtin_amdgcn_global_load_async_to_lds_b128(
            (gas1_t)(Ahi + aoff + ko),     (las3_t)(l + A_HI_OFF + aldso),     0, 0);
        __builtin_amdgcn_global_load_async_to_lds_b128(
            (gas1_t)(Ahi + aoff + ko + 8), (las3_t)(l + A_HI_OFF + aldso + 8), 0, 0);
        __builtin_amdgcn_global_load_async_to_lds_b128(
            (gas1_t)(Alo + aoff + ko),     (las3_t)(l + A_LO_OFF + aldso),     0, 0);
        __builtin_amdgcn_global_load_async_to_lds_b128(
            (gas1_t)(Alo + aoff + ko + 8), (las3_t)(l + A_LO_OFF + aldso + 8), 0, 0);
        __builtin_amdgcn_global_load_async_to_lds_b128(
            (gas1_t)(Bhi + boff + ko),     (las3_t)(l + B_HI_OFF + bldso),     0, 0);
        __builtin_amdgcn_global_load_async_to_lds_b128(
            (gas1_t)(Blo + boff + ko),     (las3_t)(l + B_LO_OFF + bldso),     0, 0);
    };
#else
    uint4 stg[6];
    auto load_stage = [&](int ks) {
        const size_t ko = (size_t)ks * KC;
        const uint4* pa0 = (const uint4*)(Ahi + aoff + ko);
        const uint4* pa1 = (const uint4*)(Alo + aoff + ko);
        stg[0] = pa0[0]; stg[1] = pa0[1];
        stg[2] = pa1[0]; stg[3] = pa1[1];
        stg[4] = *(const uint4*)(Bhi + boff + ko);
        stg[5] = *(const uint4*)(Blo + boff + ko);
    };
    auto store_stage = [&]() {
        unsigned short* l = lds[0];
        *(uint4*)(l + A_HI_OFF + aldso)     = stg[0];
        *(uint4*)(l + A_HI_OFF + aldso + 8) = stg[1];
        *(uint4*)(l + A_LO_OFF + aldso)     = stg[2];
        *(uint4*)(l + A_LO_OFF + aldso + 8) = stg[3];
        *(uint4*)(l + B_HI_OFF + bldso)     = stg[4];
        *(uint4*)(l + B_LO_OFF + bldso)     = stg[5];
    };
#endif

    v8f acc[2][2];
#pragma unroll
    for (int mt = 0; mt < 2; ++mt)
#pragma unroll
        for (int nt = 0; nt < 2; ++nt)
            acc[mt][nt] = v8f{};

    const int r  = lane & 15;           // M (A/C) or N (B) index within tile
    const int kg = lane >> 4;           // K lane-group

    // Fragment loads per documented 16-bit WMMA VGPR layouts, then bf16x3.
    auto compute = [&](const unsigned short* l) {
        U16frag afh[2], afl[2], bfh[2], bfl[2];
#pragma unroll
        for (int mt = 0; mt < 2; ++mt) {
            const int o = (wm + mt * 16 + r) * LDSS + kg * 8;   // A: K halves 0/8 and 16/24
            afh[mt].q[0] = *(const uint4*)(l + A_HI_OFF + o);
            afh[mt].q[1] = *(const uint4*)(l + A_HI_OFF + o + 16);
            afl[mt].q[0] = *(const uint4*)(l + A_LO_OFF + o);
            afl[mt].q[1] = *(const uint4*)(l + A_LO_OFF + o + 16);
        }
#pragma unroll
        for (int nt = 0; nt < 2; ++nt) {
            const int o = (wn + nt * 16 + r) * LDSS + kg * 8;   // B: K halves 0/8 and 16/24
            bfh[nt].q[0] = *(const uint4*)(l + B_HI_OFF + o);
            bfh[nt].q[1] = *(const uint4*)(l + B_HI_OFF + o + 16);
            bfl[nt].q[0] = *(const uint4*)(l + B_LO_OFF + o);
            bfl[nt].q[1] = *(const uint4*)(l + B_LO_OFF + o + 16);
        }
#pragma unroll
        for (int mt = 0; mt < 2; ++mt)
#pragma unroll
            for (int nt = 0; nt < 2; ++nt) {
                acc[mt][nt] = __builtin_amdgcn_wmma_f32_16x16x32_bf16(
                    false, afh[mt].v, false, bfh[nt].v, (short)0, acc[mt][nt], false, false);
                acc[mt][nt] = __builtin_amdgcn_wmma_f32_16x16x32_bf16(
                    false, afh[mt].v, false, bfl[nt].v, (short)0, acc[mt][nt], false, false);
                acc[mt][nt] = __builtin_amdgcn_wmma_f32_16x16x32_bf16(
                    false, afl[mt].v, false, bfh[nt].v, (short)0, acc[mt][nt], false, false);
            }
    };

#if HAVE_ASYNC_LDS
    issue_stage(0, 0);
    wait_async0();
    __syncthreads();                     // stage 0 visible to all waves
#pragma unroll 1
    for (int ks = 0; ks < KSTEPS; ++ks) {
        const int cur = ks & 1;
        if (ks + 1 < KSTEPS) issue_stage(ks + 1, cur ^ 1);  // DMA next stage
        compute(lds[cur]);
        wait_async0();                   // own DMA for next stage landed
        __syncthreads();                 // everyone done with cur + next ready
    }
#else
    load_stage(0);
#pragma unroll 1
    for (int ks = 0; ks < KSTEPS; ++ks) {
        __syncthreads();                 // previous stage fully consumed
        store_stage();
        __syncthreads();                 // tile visible to all waves
        if (ks + 1 < KSTEPS) load_stage(ks + 1);
        compute(lds[0]);
    }
#endif

    // Fused hinge epilogue. C/D layout: vgpr v, lanes 0-15 -> M=v, 16-31 -> M=v+8; N=lane&15.
    float part = 0.0f;
#pragma unroll
    for (int nt = 0; nt < 2; ++nt) {
        const int j = jBase + wn + nt * 16 + r;
        const float thr = 1.0f - diag[j];
#pragma unroll
        for (int mt = 0; mt < 2; ++mt) {
            const int ib = iBase + wm + mt * 16 + kg * 8;
#pragma unroll
            for (int v = 0; v < 8; ++v) {
                float c = thr + acc[mt][nt][v];
                c = fmaxf(c, 0.0f);
                if (ib + v == j) c = 0.0f;          // zero diagonal
                part += c;
            }
        }
    }
#pragma unroll
    for (int off = 16; off > 0; off >>= 1)
        part += __shfl_xor(part, off, 32);
    if (lane == 0)
        __hip_atomic_fetch_add(out, part, __ATOMIC_RELAXED, __HIP_MEMORY_SCOPE_AGENT);
}

// ---------------------------------------------------------------------------
extern "C" void kernel_launch(void* const* d_in, const int* in_sizes, int n_in,
                              void* d_out, int out_size, void* d_ws, size_t ws_size,
                              hipStream_t stream) {
    const float* im = (const float*)d_in[0];
    const float* s  = (const float*)d_in[1];
    const int N = in_sizes[0] / D_DIM;          // 8192
    const size_t mat = (size_t)N * D_DIM;

    // Workspace: diag (N f32) | im_hi | im_lo | s_hi | s_lo  (bf16 planes)
    float* diag = (float*)d_ws;
    unsigned short* im_hi = (unsigned short*)((char*)d_ws + (size_t)N * sizeof(float));
    unsigned short* im_lo = im_hi + mat;
    unsigned short* s_hi  = im_lo + mat;
    unsigned short* s_lo  = s_hi  + mat;

    vse_prep<<<N, 128, 0, stream>>>(im, s, im_hi, im_lo, s_hi, s_lo,
                                    diag, (float*)d_out);

    dim3 grid(N / BN, N / BM);
    vse_gemm_hinge<<<grid, 256, 0, stream>>>(im_hi, im_lo, s_hi, s_lo,
                                             diag, (float*)d_out);
}
